// MSDeformableAttention_6433861009697
// MI455X (gfx1250) — matrix operand order repaired and checked by reading, not compiled, not run on hardware
//
#include <hip/hip_runtime.h>
#include <hip/hip_bf16.h>

// ---- problem constants (match reference) ----
#define BS    2
#define NQ    21760
#define ROWS  (BS * NQ)        // 43520
#define EMBED 256
#define NH    8
#define NL    4
#define NP    4
#define HD    32               // EMBED / NH
#define KDIM  256              // all GEMMs have K = EMBED = 256
#define MTILES (ROWS / 16)     // 2720
#define WPB   8                // waves per block (256 threads, wave32)

typedef __attribute__((ext_vector_type(16))) __bf16 v16bf;
typedef __attribute__((ext_vector_type(8)))  float  v8f;

// -------------------------------------------------------------------------
// Pack fp32 weight W[K=256, N] into the per-lane bf16 B-operand layout for
// v_wmma_f32_16x16x32_bf16:
//   tile (kt, nt) -> 32 lanes x 16 halves; lane L holds column nt*16+(L&15),
//   halves i = K kt*32 + (L>>4)*16 + i.
// Flat dst index: ((kt*(N/16)+nt)*32 + lane)*16 + i   (one thread per elem)
// -------------------------------------------------------------------------
__global__ void pack_b_kernel(const float* __restrict__ W, __bf16* __restrict__ dst, int N) {
    int t = blockIdx.x * blockDim.x + threadIdx.x;
    int total = KDIM * N;
    if (t >= total) return;
    int i    = t & 15;
    int lane = (t >> 4) & 31;
    int tile = t >> 9;                 // 32*16 = 512 elems per tile
    int nTiles = N >> 4;
    int kt = tile / nTiles;
    int nt = tile - kt * nTiles;
    int k  = kt * 32 + ((lane >> 4) << 4) + i;
    int n  = nt * 16 + (lane & 15);
    dst[t] = (__bf16)W[(size_t)k * N + n];
}

// -------------------------------------------------------------------------
// C[M=43520, N] = bf16(A[M,256]) x packedB + bias   (f32 accumulate)
// One wave per (16-row M tile) x (64-col N group of 4 tiles).
// grid = (MTILES/WPB, N/64), block = 256 threads = 8 waves.
// K loop fully unrolled (8 x 32) so the scheduler can clause all A/B loads
// ahead of the 32 v_wmma_f32_16x16x32_bf16 issues.
// -------------------------------------------------------------------------
__global__ void gemm_bf16_wmma(const float* __restrict__ A,
                               const __bf16* __restrict__ Bp,
                               const float* __restrict__ bias,
                               float* __restrict__ C, int N) {
    const int lane    = threadIdx.x & 31;
    const int wv      = threadIdx.x >> 5;
    const int mTile   = blockIdx.x * WPB + wv;
    if (mTile >= MTILES) return;
    const int ntBase  = blockIdx.y * 4;
    const int halfsel = lane >> 4;     // 0: lanes 0-15, 1: lanes 16-31
    const int lrow    = lane & 15;
    const int nTiles  = N >> 4;

    // A-operand per-lane source: row (lane&15); K offset (lane>>4)*16, contiguous
    const float* arow = A + (size_t)(mTile * 16 + lrow) * KDIM + halfsel * 16;

    v8f acc[4] = {};

#pragma unroll
    for (int kt = 0; kt < 8; ++kt) {          // K = 8 * 32, fully unrolled
        const float* ap = arow + kt * 32;
        v16bf a;
#pragma unroll
        for (int i = 0; i < 8; ++i) a[i]     = (__bf16)ap[i];        // K (lane>>4)*16 ..
#pragma unroll
        for (int i = 0; i < 8; ++i) a[8 + i] = (__bf16)ap[8 + i];    // K (lane>>4)*16+8 ..

        // stage all four B tiles first -> one big load clause per k-step
        v16bf b[4];
#pragma unroll
        for (int nt = 0; nt < 4; ++nt)
            b[nt] = *(const v16bf*)(Bp +
                (((size_t)(kt * nTiles + ntBase + nt)) * 32 + lane) * 16);
#pragma unroll
        for (int nt = 0; nt < 4; ++nt)
            acc[nt] = __builtin_amdgcn_wmma_f32_16x16x32_bf16(
                false, a, false, b[nt], (short)0, acc[nt], false, false);
    }

#pragma unroll
    for (int nt = 0; nt < 4; ++nt) {
        const int col = (ntBase + nt) * 16 + lrow;
        const float bv = bias[col];
        float* crow = C + (size_t)(mTile * 16 + halfsel * 8) * N + col;
#pragma unroll
        for (int r = 0; r < 8; ++r)
            crow[(size_t)r * N] = acc[nt][r] + bv;   // row M = (lane>>4)*8 + r
    }
}

// -------------------------------------------------------------------------
// Softmax over the 16 (NL*NP) logits of each (b,q,h). One thread per group.
// -------------------------------------------------------------------------
__global__ void softmax16_kernel(float* __restrict__ aw, int total) {
    int t = blockIdx.x * blockDim.x + threadIdx.x;
    if (t >= total) return;
    float* p = aw + (size_t)t * 16;
    float m = p[0];
#pragma unroll
    for (int i = 1; i < 16; ++i) m = fmaxf(m, p[i]);
    float e[16];
    float s = 0.f;
#pragma unroll
    for (int i = 0; i < 16; ++i) { e[i] = __expf(p[i] - m); s += e[i]; }
    float inv = 1.f / s;
#pragma unroll
    for (int i = 0; i < 16; ++i) p[i] = e[i] * inv;
}

// -------------------------------------------------------------------------
// Deformable sampling. One wave per (b,q,h); lane = channel d (HD = 32).
// Each corner gather is a coalesced 128B read of vproj (L2 resident).
//   x = ref_x*W + off_x - 0.5 ; y = ref_y*H + off_y - 0.5   (normalizer cancels)
// -------------------------------------------------------------------------
__global__ void msda_sample_kernel(const float* __restrict__ vproj,
                                   const float* __restrict__ offr,
                                   const float* __restrict__ awr,
                                   const float* __restrict__ refp,
                                   float* __restrict__ samp) {
    const int Hs[NL]     = {128, 64, 32, 16};
    const int Ws[NL]     = {128, 64, 32, 16};
    const int starts[NL] = {0, 16384, 20480, 21504};

    const int lane = threadIdx.x & 31;
    const int wv   = threadIdx.x >> 5;
    const long gid = (long)blockIdx.x * WPB + wv;        // (b*NQ+q)*NH + h
    if (gid >= (long)ROWS * NH) return;
    const int  h   = (int)(gid & (NH - 1));
    const long row = gid >> 3;                           // b*NQ + q
    const int  b   = (int)(row / NQ);

    const float* offrow = offr + row * 256 + h * 32;     // ((h*4+l)*4+p)*2
    const float* awrow  = awr  + row * 128 + h * 16;     // h*16 + l*4 + p
    const float* refrow = refp + row * (NL * 2);

    float acc = 0.f;
#pragma unroll
    for (int l = 0; l < NL; ++l) {
        const int H = Hs[l], W = Ws[l], st = starts[l];
        const float rx = refrow[l * 2 + 0];
        const float ry = refrow[l * 2 + 1];
#pragma unroll
        for (int p = 0; p < NP; ++p) {
            const float x = rx * (float)W + offrow[l * 8 + p * 2 + 0] - 0.5f;
            const float y = ry * (float)H + offrow[l * 8 + p * 2 + 1] - 0.5f;
            const float a = awrow[l * 4 + p];
            const float xf = floorf(x), yf = floorf(y);
            const int   x0 = (int)xf,  y0 = (int)yf;
            const float wx1 = x - xf, wx0 = 1.f - wx1;
            const float wy1 = y - yf, wy0 = 1.f - wy1;

            float g00 = 0.f, g01 = 0.f, g10 = 0.f, g11 = 0.f;
            const long vb = ((long)b * NQ) << 8;                 // *256
            if (y0 >= 0 && y0 < H) {
                const long rbase = vb + ((long)(st + y0 * W) << 8) + h * 32 + lane;
                if (x0 >= 0     && x0 < W)     g00 = vproj[rbase + ((long)x0 << 8)];
                if (x0 + 1 >= 0 && x0 + 1 < W) g01 = vproj[rbase + ((long)(x0 + 1) << 8)];
            }
            if (y0 + 1 >= 0 && y0 + 1 < H) {
                const long rbase = vb + ((long)(st + (y0 + 1) * W) << 8) + h * 32 + lane;
                if (x0 >= 0     && x0 < W)     g10 = vproj[rbase + ((long)x0 << 8)];
                if (x0 + 1 >= 0 && x0 + 1 < W) g11 = vproj[rbase + ((long)(x0 + 1) << 8)];
            }
            acc += a * (wy0 * (wx0 * g00 + wx1 * g01) + wy1 * (wx0 * g10 + wx1 * g11));
        }
    }
    samp[row * 256 + h * 32 + lane] = acc;
}

// -------------------------------------------------------------------------
extern "C" void kernel_launch(void* const* d_in, const int* in_sizes, int n_in,
                              void* d_out, int out_size, void* d_ws, size_t ws_size,
                              hipStream_t stream) {
    (void)in_sizes; (void)n_in; (void)out_size; (void)ws_size;

    const float* query  = (const float*)d_in[0];
    const float* refp   = (const float*)d_in[1];
    const float* value  = (const float*)d_in[2];
    const float* W_off  = (const float*)d_in[3];
    const float* b_off  = (const float*)d_in[4];
    const float* W_attn = (const float*)d_in[5];
    const float* b_attn = (const float*)d_in[6];
    const float* W_v    = (const float*)d_in[7];
    const float* b_v    = (const float*)d_in[8];
    const float* W_out  = (const float*)d_in[9];
    const float* b_out  = (const float*)d_in[10];

    // workspace layout (f32 regions, then packed bf16 weights)
    float*  ws    = (float*)d_ws;
    float*  vproj = ws;                               // ROWS*256
    float*  offr  = vproj + (size_t)ROWS * 256;       // ROWS*256
    float*  awr   = offr  + (size_t)ROWS * 256;       // ROWS*128
    float*  samp  = awr   + (size_t)ROWS * 128;       // ROWS*256
    __bf16* pWv   = (__bf16*)(samp + (size_t)ROWS * 256);
    __bf16* pWoff = pWv   + 256 * 256;
    __bf16* pWatt = pWoff + 256 * 256;
    __bf16* pWout = pWatt + 256 * 128;

    // 1) pack weights to WMMA B-operand layout (bf16)
    pack_b_kernel<<<(256 * 256 + 255) / 256, 256, 0, stream>>>(W_v,    pWv,   256);
    pack_b_kernel<<<(256 * 256 + 255) / 256, 256, 0, stream>>>(W_off,  pWoff, 256);
    pack_b_kernel<<<(256 * 128 + 255) / 256, 256, 0, stream>>>(W_attn, pWatt, 128);
    pack_b_kernel<<<(256 * 256 + 255) / 256, 256, 0, stream>>>(W_out,  pWout, 256);

    // 2) input GEMMs
    dim3 blk(256);
    gemm_bf16_wmma<<<dim3(MTILES / WPB, 4), blk, 0, stream>>>(value, pWv,   b_v,    vproj, 256);
    gemm_bf16_wmma<<<dim3(MTILES / WPB, 4), blk, 0, stream>>>(query, pWoff, b_off,  offr,  256);
    gemm_bf16_wmma<<<dim3(MTILES / WPB, 2), blk, 0, stream>>>(query, pWatt, b_attn, awr,   128);

    // 3) softmax over NL*NP=16 per (b,q,h)
    const int groups = ROWS * NH;                     // 348160
    softmax16_kernel<<<(groups + 255) / 256, 256, 0, stream>>>(awr, groups);

    // 4) deformable bilinear sampling (one wave per (b,q,h))
    msda_sample_kernel<<<(groups + WPB - 1) / WPB, 256, 0, stream>>>(vproj, offr, awr, refp, samp);

    // 5) output GEMM straight into d_out
    gemm_bf16_wmma<<<dim3(MTILES / WPB, 4), blk, 0, stream>>>(samp, pWout, b_out, (float*)d_out, 256);
}
